// Syntax_Transformer_BERTModel_50852412785300
// MI455X (gfx1250) — matrix operand order repaired and checked by reading, not compiled, not run on hardware
//
#include <hip/hip_runtime.h>
#include <hip/hip_bf16.h>

// Sizes from the reference
#define B_   2
#define S_   128
#define H_   768
#define DE_  128
#define HE_  4
#define HT_  12
#define L_   2
#define DEH_ 32
#define HTH_ 64

typedef __attribute__((ext_vector_type(16))) _Float16 v16h;
typedef __attribute__((ext_vector_type(8)))  _Float16 v8h;
typedef __attribute__((ext_vector_type(8)))  float    v8f;

// ---------------------------------------------------------------------------
// WMMA fragment helpers (CDNA5 16x16x32 f16, wave32)
// A-frag (16-bit, 16x32): lane<16 holds row (lane), K k0..k0+15 contiguous;
//                         lane>=16 holds row (lane-16), K k0+16..k0+31
// B-frag: lane<16 holds col (lane), K k0..k0+15 contiguous; lane>=16 same col
//         range with K k0+16..k0+31  -> load from B^T ([N][K]) contiguously.
// C/D: VGPR r -> row r + 8*(lane>=16), col = lane&15.
// ---------------------------------------------------------------------------
__device__ __forceinline__ v16h load_afrag(const _Float16* rowp, int k0, int lhi) {
  return *(const v16h*)(rowp + k0 + lhi * 16);
}

__device__ __forceinline__ v8f wmma_f16(v16h a, v16h b, v8f c) {
  return __builtin_amdgcn_wmma_f32_16x16x32_f16(false, a, false, b, (short)0, c,
                                                false, false);
}

// ---------------------------------------------------------------------------
// Prep kernels
// ---------------------------------------------------------------------------
__global__ void k_transpose_h(const float* __restrict__ W, _Float16* __restrict__ Bt,
                              int K, int N) {
  int e = blockIdx.x * blockDim.x + threadIdx.x;
  if (e >= K * N) return;
  int n = e / K, k = e - n * K;
  Bt[e] = (_Float16)W[(size_t)k * N + n];
}

__global__ void k_cvt_h(const float* __restrict__ x, _Float16* __restrict__ y, int n) {
  int e = blockIdx.x * blockDim.x + threadIdx.x;
  if (e < n) y[e] = (_Float16)x[e];
}

// edge_emb (f16) = dep_table[edge_ids]   ((B*S*S) x DE)
__global__ void k_gather(const int* __restrict__ ids, const float* __restrict__ table,
                         _Float16* __restrict__ out) {
  size_t e = (size_t)blockIdx.x * blockDim.x + threadIdx.x;
  int d = (int)(e & (DE_ - 1));
  size_t n = e >> 7;
  out[e] = (_Float16)table[(size_t)ids[n] * DE_ + d];
}

// ---------------------------------------------------------------------------
// Generic WMMA GEMM: C[M,N] = A[M,K] * Bt[N,K]^T + bias ; fp32 and/or f16 out.
// block = 256 (8 waves), wave w -> 16x16 tile at (blockIdx.x*16, blockIdx.y*128+w*16)
// M%16==0, N%16==0 (grid.y covers N in 128-chunks), K%32==0.
// ---------------------------------------------------------------------------
__global__ void k_gemm_wmma(const _Float16* __restrict__ A, const _Float16* __restrict__ Bt,
                            const float* __restrict__ bias, float* __restrict__ Cf,
                            _Float16* __restrict__ Ch, int M, int N, int K) {
  int wave = threadIdx.x >> 5;
  int lane = threadIdx.x & 31;
  int l15 = lane & 15, lhi = lane >> 4;
  int r0 = blockIdx.x * 16;
  int n0 = blockIdx.y * 128 + wave * 16;
  if (n0 >= N) return;
  const _Float16* ap = A + (size_t)(r0 + l15) * K;
  const _Float16* bp = Bt + (size_t)(n0 + l15) * K;
  v8f acc = {};
  for (int k0 = 0; k0 < K; k0 += 32) {
    v16h a = load_afrag(ap, k0, lhi);
    v16h b = *(const v16h*)(bp + k0 + lhi * 16);
    acc = wmma_f16(a, b, acc);
  }
  int col = n0 + l15;
  float bv = bias ? bias[col] : 0.0f;
  int rowb = r0 + lhi * 8;
#pragma unroll
  for (int r = 0; r < 8; ++r) {
    float v = acc[r] + bv;
    size_t idx = (size_t)(rowb + r) * N + col;
    if (Cf) Cf[idx] = v;
    if (Ch) Ch[idx] = (_Float16)v;
  }
}

// ---------------------------------------------------------------------------
// Dynamic edge attention: one workgroup per (b,h,i).
//   S=128 rows(j) x 128 cols(k), head dim 32.
//   Phase1: wave w computes score rows [16w,16w+16) via 8 WMMAs (regs only),
//           row mask + softmax with 16-lane shfl reductions, f16 probs -> LDS.
//   Phase2: probs(128x128) @ v(128x32) via WMMA, vT staged in LDS.
//   merged written in reference "full" layout [(b*S+i)*S+j][h*32+d] (fp32).
// ---------------------------------------------------------------------------
__global__ void k_dyn_attn(const _Float16* __restrict__ qh, const _Float16* __restrict__ kh,
                           const _Float16* __restrict__ vh, const int* __restrict__ dep_mask,
                           float* __restrict__ merged) {
  __shared__ _Float16 probs[S_ * S_];  // 32 KB
  __shared__ _Float16 vT[DEH_ * S_];   // 8 KB
  int g = blockIdx.x;
  int i = g & (S_ - 1);
  int h = (g >> 7) & (HE_ - 1);
  int b = g >> 9;
  int tid = threadIdx.x;
  int lane = tid & 31, wave = tid >> 5;
  int l15 = lane & 15, lhi = lane >> 4;
  size_t base = (((size_t)(b * S_ + i)) * S_) * DE_ + h * DEH_;  // + j*DE_ + d

  // stage v^T : vT[n][k] = v[k][n]
  for (int e = tid; e < DEH_ * S_; e += 256) {
    int n = e >> 7, k2 = e & (S_ - 1);
    vT[n * S_ + k2] = vh[base + (size_t)k2 * DE_ + n];
  }
  __syncthreads();

  // ---- phase 1: scores rows [jr, jr+16), all 128 cols -------------------
  int jr = 16 * wave;
  v8f acc[8] = {};
  v16h a = load_afrag(qh + base + (size_t)(jr + l15) * DE_, 0, lhi);
#pragma unroll
  for (int c = 0; c < 8; ++c) {
    v16h bb = *(const v16h*)(kh + base + (size_t)(16 * c + l15) * DE_ + lhi * 16);
    acc[c] = wmma_f16(a, bb, acc[c]);
  }
  // row mask: dep_mask[b,i,j] == 0 -> whole row -10000 (softmax -> uniform)
  const int* dm = dep_mask + (size_t)(b * S_ + i) * S_;
#pragma unroll
  for (int r = 0; r < 8; ++r) {
    if (dm[jr + r + 8 * lhi] == 0) {
#pragma unroll
      for (int c = 0; c < 8; ++c) acc[c][r] = -10000.0f;
    }
  }
  // softmax per row; attn = softmax(wgt)/sqrt(32) (scale AFTER softmax, per ref)
  const float inv_sq = 0.17677669529663687f;  // 1/sqrt(32)
#pragma unroll
  for (int r = 0; r < 8; ++r) {
    float m = -3.4e38f;
#pragma unroll
    for (int c = 0; c < 8; ++c) m = fmaxf(m, acc[c][r]);
    for (int off = 1; off < 16; off <<= 1) m = fmaxf(m, __shfl_xor(m, off, 32));
    float ex[8], sum = 0.0f;
#pragma unroll
    for (int c = 0; c < 8; ++c) { ex[c] = __expf(acc[c][r] - m); sum += ex[c]; }
    for (int off = 1; off < 16; off <<= 1) sum += __shfl_xor(sum, off, 32);
    float sc = inv_sq / sum;
    int row = jr + r + 8 * lhi;
#pragma unroll
    for (int c = 0; c < 8; ++c)
      probs[row * S_ + 16 * c + l15] = (_Float16)(ex[c] * sc);
  }
  __syncthreads();

  // ---- phase 2: merged rows [jr, jr+16), 32 cols ------------------------
  v8f m0 = {}, m1 = {};
  for (int k0 = 0; k0 < S_; k0 += 32) {
    v16h pa = load_afrag(probs + (jr + l15) * S_, k0, lhi);
    v16h b0 = *(const v16h*)(vT + (0 + l15) * S_ + k0 + lhi * 16);
    v16h b1 = *(const v16h*)(vT + (16 + l15) * S_ + k0 + lhi * 16);
    m0 = wmma_f16(pa, b0, m0);
    m1 = wmma_f16(pa, b1, m1);
  }
#pragma unroll
  for (int r = 0; r < 8; ++r) {
    int row = jr + r + 8 * lhi;
    merged[base + (size_t)row * DE_ + l15] = m0[r];
    merged[base + (size_t)row * DE_ + 16 + l15] = m1[r];
  }
}

// ---------------------------------------------------------------------------
// Gate: alph = sigmoid([m, mT] @ aw + ab); edge_feat = (1-alph)*m + alph*mT
// block per (b,i,j), 128 threads (one per d).
// ---------------------------------------------------------------------------
__global__ void k_gate(const float* __restrict__ merged, const float* __restrict__ aw,
                       const float* __restrict__ ab, float* __restrict__ efeat) {
  __shared__ float red[128];
  int g = blockIdx.x;
  int j = g & (S_ - 1);
  int i = (g >> 7) & (S_ - 1);
  int b = g >> 14;
  size_t idx1 = (size_t)g * DE_;                                   // m[b,i,j]
  size_t idx2 = (((size_t)(b * S_ + j)) * S_ + i) * DE_;           // m[b,j,i]
  int d = threadIdx.x;
  float m1 = merged[idx1 + d], m2 = merged[idx2 + d];
  red[d] = m1 * aw[d] + m2 * aw[DE_ + d];
  __syncthreads();
  for (int off = 64; off; off >>= 1) {
    if (d < off) red[d] += red[d + off];
    __syncthreads();
  }
  float al = 1.0f / (1.0f + __expf(-(red[0] + ab[0])));
  efeat[idx1 + d] = (1.0f - al) * m1 + al * m2;
}

// ---------------------------------------------------------------------------
// qe[(b,h,i),d] = ekw[d, h*64:+64] . q[b,h,i,:]   (algebraic refactor of q.ek)
// qeb[(b,h,i)]  = q[b,h,i,:] . ekb[h*64:+64]
// ---------------------------------------------------------------------------
__global__ void k_qe(const float* __restrict__ q, const float* __restrict__ ekw,
                     const float* __restrict__ ekb, float* __restrict__ qe,
                     float* __restrict__ qeb) {
  int gid = blockIdx.x * blockDim.x + threadIdx.x;
  if (gid >= B_ * HT_ * S_ * DE_) return;
  int d = gid & (DE_ - 1);
  int ridx = gid >> 7;
  int i = ridx & (S_ - 1);
  int h = (ridx >> 7) % HT_;
  int b = ridx / (HT_ * S_);
  const float* qrow = q + ((size_t)(b * S_ + i)) * H_ + h * HTH_;
  const float* wcol = ekw + (size_t)d * H_ + h * HTH_;
  float s = 0.0f;
#pragma unroll 8
  for (int t = 0; t < HTH_; ++t) s += qrow[t] * wcol[t];
  qe[(size_t)ridx * DE_ + d] = s;
  if (d == 0) {
    const float* eb = ekb + h * HTH_;
    float sb = 0.0f;
#pragma unroll 8
    for (int t = 0; t < HTH_; ++t) sb += qrow[t] * eb[t];
    qeb[ridx] = sb;
  }
}

// ---------------------------------------------------------------------------
// Syntax attention, block per (b,h,i), 128 threads (thread = j).
// scores_j = (q.kt_j + WE*(ef_ij.qe + qeb)) / 8 ; mask -> -10000 ; softmax;
// ctx = probs@vt + WE*((sum_j p_j*ef_ij)@evw_h + evb_h)
// ---------------------------------------------------------------------------
__global__ void k_syn_attn(const float* __restrict__ qt, const float* __restrict__ kt,
                           const float* __restrict__ vt, const float* __restrict__ efeat,
                           const float* __restrict__ qe, const float* __restrict__ qeb,
                           const int* __restrict__ dep_mask, const float* __restrict__ evw,
                           const float* __restrict__ evb, float* __restrict__ ctx) {
  __shared__ float red[128];
  __shared__ float probs[128];
  __shared__ float pe[128];
  int g = blockIdx.x;
  int i = g & (S_ - 1);
  int h = (g >> 7) % HT_;
  int b = (g >> 7) / HT_;
  int j = threadIdx.x;
  int ridx = (b * HT_ + h) * S_ + i;
  const float* qrow = qt + ((size_t)(b * S_ + i)) * H_ + h * HTH_;
  const float* krow = kt + ((size_t)(b * S_ + j)) * H_ + h * HTH_;
  float s1 = 0.0f;
#pragma unroll 8
  for (int t = 0; t < HTH_; ++t) s1 += qrow[t] * krow[t];
  const float* ef = efeat + (((size_t)(b * S_ + i)) * S_ + j) * DE_;
  const float* qer = qe + (size_t)ridx * DE_;
  float s2 = 0.0f;
#pragma unroll 8
  for (int d = 0; d < DE_; ++d) s2 += ef[d] * qer[d];
  float s = (s1 + 0.5f * (s2 + qeb[ridx])) * 0.125f;  // /sqrt(64)
  if (dep_mask[((size_t)(b * S_ + i)) * S_ + j] == 0) s = -10000.0f;
  red[j] = s;
  __syncthreads();
  for (int off = 64; off; off >>= 1) {
    if (j < off) red[j] = fmaxf(red[j], red[j + off]);
    __syncthreads();
  }
  float m = red[0];
  __syncthreads();
  float e = __expf(s - m);
  red[j] = e;
  __syncthreads();
  for (int off = 64; off; off >>= 1) {
    if (j < off) red[j] += red[j + off];
    __syncthreads();
  }
  probs[j] = e / red[0];
  __syncthreads();
  // pe[d] = sum_j probs_j * ef[b,i,j,d]  (thread doubles as d)
  const float* efb = efeat + ((size_t)(b * S_ + i)) * S_ * DE_;
  float acc = 0.0f;
  for (int j2 = 0; j2 < S_; ++j2) acc += probs[j2] * efb[(size_t)j2 * DE_ + j];
  pe[j] = acc;
  __syncthreads();
  if (j < HTH_) {
    int t = j;
    float c = 0.0f;
    for (int j2 = 0; j2 < S_; ++j2)
      c += probs[j2] * vt[((size_t)(b * S_ + j2)) * H_ + h * HTH_ + t];
    float e2 = 0.0f;
    for (int d = 0; d < DE_; ++d) e2 += pe[d] * evw[(size_t)d * H_ + h * HTH_ + t];
    c += 0.5f * (e2 + evb[h * HTH_ + t]);
    ctx[((size_t)(b * S_ + i)) * H_ + h * HTH_ + t] = c;
  }
}

// ---------------------------------------------------------------------------
// Residual + LayerNorm over H=768; block per (b,i), 256 threads (3 elems each)
// ---------------------------------------------------------------------------
__global__ void k_ln(const float* __restrict__ tok, const float* __restrict__ ctx,
                     const float* __restrict__ gg, const float* __restrict__ be,
                     float* __restrict__ out) {
  __shared__ float red[256];
  int row = blockIdx.x, t = threadIdx.x;
  size_t rb = (size_t)row * H_;
  float x0 = tok[rb + t] + ctx[rb + t];
  float x1 = tok[rb + t + 256] + ctx[rb + t + 256];
  float x2 = tok[rb + t + 512] + ctx[rb + t + 512];
  red[t] = x0 + x1 + x2;
  __syncthreads();
  for (int off = 128; off; off >>= 1) {
    if (t < off) red[t] += red[t + off];
    __syncthreads();
  }
  float mu = red[0] * (1.0f / H_);
  __syncthreads();
  float d0 = x0 - mu, d1 = x1 - mu, d2 = x2 - mu;
  red[t] = d0 * d0 + d1 * d1 + d2 * d2;
  __syncthreads();
  for (int off = 128; off; off >>= 1) {
    if (t < off) red[t] += red[t + off];
    __syncthreads();
  }
  float inv = rsqrtf(red[0] * (1.0f / H_) + 1e-5f);
  out[rb + t] = d0 * inv * gg[t] + be[t];
  out[rb + t + 256] = d1 * inv * gg[t + 256] + be[t + 256];
  out[rb + t + 512] = d2 * inv * gg[t + 512] + be[t + 512];
}

// ---------------------------------------------------------------------------
extern "C" void kernel_launch(void* const* d_in, const int* in_sizes, int n_in,
                              void* d_out, int out_size, void* d_ws, size_t ws_size,
                              hipStream_t stream) {
  (void)in_sizes; (void)n_in; (void)out_size; (void)ws_size;
  const float* token     = (const float*)d_in[0];
  const int*   edge_ids  = (const int*)d_in[1];
  const int*   dep_mask  = (const int*)d_in[2];
  const float* dep_table = (const float*)d_in[3];
  const float* dl_wq = (const float*)d_in[4];
  const float* dl_bq = (const float*)d_in[5];
  const float* dl_wk = (const float*)d_in[6];
  const float* dl_bk = (const float*)d_in[7];
  const float* dl_wv = (const float*)d_in[8];
  const float* dl_bv = (const float*)d_in[9];
  const float* dl_aw = (const float*)d_in[10];
  const float* dl_ab = (const float*)d_in[11];
  const float* st_wq = (const float*)d_in[12];
  const float* st_bq = (const float*)d_in[13];
  const float* st_wk = (const float*)d_in[14];
  const float* st_bk = (const float*)d_in[15];
  const float* st_wv = (const float*)d_in[16];
  const float* st_bv = (const float*)d_in[17];
  const float* st_ekw = (const float*)d_in[18];
  const float* st_ekb = (const float*)d_in[19];
  const float* st_evw = (const float*)d_in[20];
  const float* st_evb = (const float*)d_in[21];
  const float* st_lng = (const float*)d_in[22];
  const float* st_lnb = (const float*)d_in[23];

  const size_t NE = (size_t)B_ * S_ * S_;          // 32768 edge rows
  const size_t NT = (size_t)B_ * S_;               // 256 token rows

  char* w = (char*)d_ws;
  auto alloc = [&](size_t bytes) -> char* {
    char* p = w;
    w += (bytes + 255) & ~(size_t)255;
    return p;
  };
  _Float16* dlq_t = (_Float16*)alloc(DE_ * DE_ * 2);
  _Float16* dlk_t = (_Float16*)alloc(DE_ * DE_ * 2);
  _Float16* dlv_t = (_Float16*)alloc(DE_ * DE_ * 2);
  _Float16* stq_t = (_Float16*)alloc((size_t)H_ * H_ * 2);
  _Float16* stk_t = (_Float16*)alloc((size_t)H_ * H_ * 2);
  _Float16* stv_t = (_Float16*)alloc((size_t)H_ * H_ * 2);
  _Float16* emb_h = (_Float16*)alloc(NE * DE_ * 2);
  _Float16* qh    = (_Float16*)alloc(NE * DE_ * 2);
  _Float16* kh    = (_Float16*)alloc(NE * DE_ * 2);
  _Float16* vh    = (_Float16*)alloc(NE * DE_ * 2);
  float* merged   = (float*)alloc(NE * DE_ * 4);
  float* efeat    = (float*)alloc(NE * DE_ * 4);
  _Float16* tok_h = (_Float16*)alloc(NT * H_ * 2);
  float* qt   = (float*)alloc(NT * H_ * 4);
  float* kt   = (float*)alloc(NT * H_ * 4);
  float* vt   = (float*)alloc(NT * H_ * 4);
  float* qe   = (float*)alloc((size_t)B_ * HT_ * S_ * DE_ * 4);
  float* qeb  = (float*)alloc((size_t)B_ * HT_ * S_ * 4);
  float* ctxb = (float*)alloc(NT * H_ * 4);
  float* tokb = (float*)alloc(NT * H_ * 4);

  // ---- dynamic layer -----------------------------------------------------
  k_transpose_h<<<64, 256, 0, stream>>>(dl_wq, dlq_t, DE_, DE_);
  k_transpose_h<<<64, 256, 0, stream>>>(dl_wk, dlk_t, DE_, DE_);
  k_transpose_h<<<64, 256, 0, stream>>>(dl_wv, dlv_t, DE_, DE_);
  k_gather<<<(int)(NE * DE_ / 256), 256, 0, stream>>>(edge_ids, dep_table, emb_h);

  dim3 gd((int)(NE / 16), 1);  // N=128 -> one 128-col block
  k_gemm_wmma<<<gd, 256, 0, stream>>>(emb_h, dlq_t, dl_bq, nullptr, qh, (int)NE, DE_, DE_);
  k_gemm_wmma<<<gd, 256, 0, stream>>>(emb_h, dlk_t, dl_bk, nullptr, kh, (int)NE, DE_, DE_);
  k_gemm_wmma<<<gd, 256, 0, stream>>>(emb_h, dlv_t, dl_bv, nullptr, vh, (int)NE, DE_, DE_);

  k_dyn_attn<<<B_ * HE_ * S_, 256, 0, stream>>>(qh, kh, vh, dep_mask, merged);
  k_gate<<<(int)NE, 128, 0, stream>>>(merged, dl_aw, dl_ab, efeat);

  // ---- syntax layers -----------------------------------------------------
  for (int l = 0; l < L_; ++l) {
    const float* tsrc = (l == 0) ? token : tokb;
    float* tdst = (l == L_ - 1) ? (float*)d_out : tokb;
    k_cvt_h<<<(int)(NT * H_ + 255) / 256, 256, 0, stream>>>(tsrc, tok_h, (int)(NT * H_));
    k_transpose_h<<<(H_ * H_ + 255) / 256, 256, 0, stream>>>(st_wq + (size_t)l * H_ * H_, stq_t, H_, H_);
    k_transpose_h<<<(H_ * H_ + 255) / 256, 256, 0, stream>>>(st_wk + (size_t)l * H_ * H_, stk_t, H_, H_);
    k_transpose_h<<<(H_ * H_ + 255) / 256, 256, 0, stream>>>(st_wv + (size_t)l * H_ * H_, stv_t, H_, H_);

    dim3 gt((int)(NT / 16), H_ / 128);
    k_gemm_wmma<<<gt, 256, 0, stream>>>(tok_h, stq_t, st_bq + l * H_, qt, nullptr, (int)NT, H_, H_);
    k_gemm_wmma<<<gt, 256, 0, stream>>>(tok_h, stk_t, st_bk + l * H_, kt, nullptr, (int)NT, H_, H_);
    k_gemm_wmma<<<gt, 256, 0, stream>>>(tok_h, stv_t, st_bv + l * H_, vt, nullptr, (int)NT, H_, H_);

    k_qe<<<(B_ * HT_ * S_ * DE_) / 256, 256, 0, stream>>>(
        qt, st_ekw + (size_t)l * DE_ * H_, st_ekb + l * H_, qe, qeb);
    k_syn_attn<<<B_ * HT_ * S_, 128, 0, stream>>>(
        qt, kt, vt, efeat, qe, qeb, dep_mask,
        st_evw + (size_t)l * DE_ * H_, st_evb + l * H_, ctxb);
    k_ln<<<(int)NT, 256, 0, stream>>>(tsrc, ctxb, st_lng + l * H_, st_lnb + l * H_, tdst);
  }
}
